// PatchedSelfAttention_68968584839838
// MI455X (gfx1250) — matrix operand order repaired and checked by reading, not compiled, not run on hardware
//
#include <hip/hip_runtime.h>
#include <hip/hip_bf16.h>

typedef __bf16 bf16_t;
typedef __attribute__((ext_vector_type(16))) __bf16 v16bf;
typedef __attribute__((ext_vector_type(8)))  __bf16 v8bf;
typedef __attribute__((ext_vector_type(8)))  float  v8f;

union V16u { v16bf v; v8bf h[2]; };

__device__ __forceinline__ bf16_t f2bf(float f) {
    unsigned u = __builtin_bit_cast(unsigned, f);
    unsigned r = (u + 0x7FFFu + ((u >> 16) & 1u)) >> 16;
    return __builtin_bit_cast(bf16_t, (unsigned short)r);
}

// Pack two floats into two bf16 in one dword (RNE).
__device__ __forceinline__ unsigned pk2(float lo, float hi) {
#if __has_builtin(__builtin_amdgcn_cvt_pk_bf16_f32)
    typedef __attribute__((ext_vector_type(2))) __bf16 v2bf;
    v2bf p = __builtin_amdgcn_cvt_pk_bf16_f32(lo, hi);
    return __builtin_bit_cast(unsigned, p);
#else
    unsigned a = __builtin_bit_cast(unsigned, lo);
    unsigned b = __builtin_bit_cast(unsigned, hi);
    a = (a + 0x7FFFu + ((a >> 16) & 1u)) >> 16;
    b = (b + 0x7FFFu + ((b >> 16) & 1u)) >> 16;
    return a | (b << 16);
#endif
}

// Async 16-byte global -> LDS copy (per active lane), tracked by ASYNCcnt.
__device__ __forceinline__ void async_ld_b128(unsigned lds_addr, const void* gaddr) {
    asm volatile("global_load_async_to_lds_b128 %0, %1, off"
                 :: "v"(lds_addr), "v"(gaddr) : "memory");
}
__device__ __forceinline__ void wait_async0() {
    asm volatile("s_wait_asynccnt 0" ::: "memory");
}

// ---------------------------------------------------------------------------
// C[M,N] = A[M,K] (f32) * W[N,K]^T (f32) + bias, bf16 WMMA, f32 accumulate.
// Block 256 threads (8 waves). Tile 128x128, BK=32. Wave tile 32x64.
// ---------------------------------------------------------------------------
__global__ __launch_bounds__(256) void gemm_bf16(const float* __restrict__ A,
                                                 const float* __restrict__ W,
                                                 const float* __restrict__ bias,
                                                 float* __restrict__ C,
                                                 int M, int N, int K)
{
    __shared__ bf16_t As[128 * 32];
    __shared__ bf16_t Ws[128 * 32];
    const int bm   = blockIdx.y * 128;
    const int bn   = blockIdx.x * 128;
    const int tid  = threadIdx.x;
    const int wave = tid >> 5;
    const int lane = tid & 31;
    const int lh   = lane >> 4;
    const int ln   = lane & 15;
    const int wm   = wave & 3;
    const int wn   = wave >> 2;

    v8f acc[2][4];
    for (int t = 0; t < 2; ++t)
        for (int n = 0; n < 4; ++n)
            for (int r = 0; r < 8; ++r) acc[t][n][r] = 0.0f;

    const int lrow = tid >> 1;
    const int lcol = (tid & 1) * 16;
    const float* gA = A + (size_t)(bm + lrow) * K + lcol;
    const float* gW = W + (size_t)(bn + lrow) * K + lcol;

    for (int k0 = 0; k0 < K; k0 += 32) {
        if (k0 + 32 < K) {
            __builtin_prefetch(gA + k0 + 32, 0, 3);
            __builtin_prefetch(gW + k0 + 32, 0, 3);
        }
        __syncthreads();
        const float4* ga = (const float4*)(gA + k0);
        const float4* gw = (const float4*)(gW + k0);
        uint4 ap0, ap1, wp0, wp1;
        {
            float4 a0 = ga[0], a1 = ga[1], a2 = ga[2], a3 = ga[3];
            float4 w0 = gw[0], w1 = gw[1], w2 = gw[2], w3 = gw[3];
            ap0.x = pk2(a0.x, a0.y); ap0.y = pk2(a0.z, a0.w);
            ap0.z = pk2(a1.x, a1.y); ap0.w = pk2(a1.z, a1.w);
            ap1.x = pk2(a2.x, a2.y); ap1.y = pk2(a2.z, a2.w);
            ap1.z = pk2(a3.x, a3.y); ap1.w = pk2(a3.z, a3.w);
            wp0.x = pk2(w0.x, w0.y); wp0.y = pk2(w0.z, w0.w);
            wp0.z = pk2(w1.x, w1.y); wp0.w = pk2(w1.z, w1.w);
            wp1.x = pk2(w2.x, w2.y); wp1.y = pk2(w2.z, w2.w);
            wp1.z = pk2(w3.x, w3.y); wp1.w = pk2(w3.z, w3.w);
        }
        uint4* da = (uint4*)(As + lrow * 32 + lcol);
        uint4* dw = (uint4*)(Ws + lrow * 32 + lcol);
        da[0] = ap0; da[1] = ap1;
        dw[0] = wp0; dw[1] = wp1;
        __syncthreads();

        V16u af[2], bw[4];
        for (int t = 0; t < 2; ++t) {
            const bf16_t* pa = As + (wm * 32 + t * 16 + ln) * 32;
            af[t].h[0] = *(const v8bf*)(pa + lh * 8);
            af[t].h[1] = *(const v8bf*)(pa + 16 + lh * 8);
        }
        for (int n = 0; n < 4; ++n) {
            const bf16_t* pw = Ws + (wn * 64 + n * 16 + ln) * 32;
            bw[n].h[0] = *(const v8bf*)(pw + lh * 8);
            bw[n].h[1] = *(const v8bf*)(pw + 16 + lh * 8);
        }
        for (int t = 0; t < 2; ++t)
            for (int n = 0; n < 4; ++n)
                acc[t][n] = __builtin_amdgcn_wmma_f32_16x16x32_bf16(
                    false, af[t].v, false, bw[n].v, (short)0, acc[t][n], false, false);
    }

    for (int t = 0; t < 2; ++t)
        for (int n = 0; n < 4; ++n) {
            int col = bn + wn * 64 + n * 16 + ln;
            float bsc = bias ? bias[col] : 0.0f;
            float* cp = C + (size_t)(bm + wm * 32 + t * 16 + 8 * lh) * N + col;
            for (int r = 0; r < 8; ++r)
                cp[(size_t)r * N] = acc[t][n][r] + bsc;
        }
}

// ---------------------------------------------------------------------------
// Same GEMM but A is already bf16: A tile staged with async global->LDS b128.
// ---------------------------------------------------------------------------
__global__ __launch_bounds__(256) void gemm_abf16(const bf16_t* __restrict__ A,
                                                  const float* __restrict__ W,
                                                  float* __restrict__ C,
                                                  int M, int N, int K)
{
    __shared__ bf16_t As[128 * 32];
    __shared__ bf16_t Ws[128 * 32];
    const int bm   = blockIdx.y * 128;
    const int bn   = blockIdx.x * 128;
    const int tid  = threadIdx.x;
    const int wave = tid >> 5;
    const int lane = tid & 31;
    const int lh   = lane >> 4;
    const int ln   = lane & 15;
    const int wm   = wave & 3;
    const int wn   = wave >> 2;

    v8f acc[2][4];
    for (int t = 0; t < 2; ++t)
        for (int n = 0; n < 4; ++n)
            for (int r = 0; r < 8; ++r) acc[t][n][r] = 0.0f;

    const int lrow = tid >> 1;
    const int lcol = (tid & 1) * 16;
    const bf16_t* gA = A + (size_t)(bm + lrow) * K + lcol;
    const float*  gW = W + (size_t)(bn + lrow) * K + lcol;
    const unsigned la = (unsigned)(size_t)(As + lrow * 32 + lcol);

    for (int k0 = 0; k0 < K; k0 += 32) {
        if (k0 + 32 < K) {
            __builtin_prefetch(gA + k0 + 32, 0, 3);
            __builtin_prefetch(gW + k0 + 32, 0, 3);
        }
        __syncthreads();
        // A: direct async copy (no conversion needed)
        async_ld_b128(la, gA + k0);
        async_ld_b128(la + 16, gA + k0 + 8);
        // W: f32 -> bf16 pack
        const float4* gw = (const float4*)(gW + k0);
        uint4 wp0, wp1;
        {
            float4 w0 = gw[0], w1 = gw[1], w2 = gw[2], w3 = gw[3];
            wp0.x = pk2(w0.x, w0.y); wp0.y = pk2(w0.z, w0.w);
            wp0.z = pk2(w1.x, w1.y); wp0.w = pk2(w1.z, w1.w);
            wp1.x = pk2(w2.x, w2.y); wp1.y = pk2(w2.z, w2.w);
            wp1.z = pk2(w3.x, w3.y); wp1.w = pk2(w3.z, w3.w);
        }
        uint4* dw = (uint4*)(Ws + lrow * 32 + lcol);
        dw[0] = wp0; dw[1] = wp1;
        wait_async0();
        __syncthreads();

        V16u af[2], bw[4];
        for (int t = 0; t < 2; ++t) {
            const bf16_t* pa = As + (wm * 32 + t * 16 + ln) * 32;
            af[t].h[0] = *(const v8bf*)(pa + lh * 8);
            af[t].h[1] = *(const v8bf*)(pa + 16 + lh * 8);
        }
        for (int n = 0; n < 4; ++n) {
            const bf16_t* pw = Ws + (wn * 64 + n * 16 + ln) * 32;
            bw[n].h[0] = *(const v8bf*)(pw + lh * 8);
            bw[n].h[1] = *(const v8bf*)(pw + 16 + lh * 8);
        }
        for (int t = 0; t < 2; ++t)
            for (int n = 0; n < 4; ++n)
                acc[t][n] = __builtin_amdgcn_wmma_f32_16x16x32_bf16(
                    false, af[t].v, false, bw[n].v, (short)0, acc[t][n], false, false);
    }

    for (int t = 0; t < 2; ++t)
        for (int n = 0; n < 4; ++n) {
            int col = bn + wn * 64 + n * 16 + ln;
            float* cp = C + (size_t)(bm + wm * 32 + t * 16 + 8 * lh) * N + col;
            for (int r = 0; r < 8; ++r)
                cp[(size_t)r * N] = acc[t][n][r];
        }
}

// ---------------------------------------------------------------------------
// RoPE + reorder: mixed[SQ,4608] -> qbf [B][2][16][1024][128] (scaled 1/sqrt(128)),
// new K (rope'd) -> kbf [B][2][2048][128] at s>=1024,
// new V          -> vtb [B][2][128][2048] (TRANSPOSED) at s>=1024.
// ---------------------------------------------------------------------------
__global__ __launch_bounds__(128) void prep_qkv(const float* __restrict__ mixed,
                                                const float* __restrict__ rope,
                                                bf16_t* __restrict__ qbf,
                                                bf16_t* __restrict__ kbf,
                                                bf16_t* __restrict__ vtb)
{
    const int sq  = blockIdx.x;
    const int tid = threadIdx.x;
    const int b   = sq >> 10;
    const int qq  = sq & 1023;
    const float scale = 0.08838834764831845f;  // 1/sqrt(128)
    const float* mrow = mixed + (size_t)sq * 4608;
    const float* rrow = rope + (size_t)sq * 64;

    for (int i = 0; i < 16; ++i) {
        int u  = tid + i * 128;
        int h  = u >> 6;
        int u6 = u & 63;
        const float* base = mrow + h * 128;
        float y0, y1; int d0;
        if (u6 < 32) {
            d0 = u6 * 2;
            float c = rrow[u6 * 2], s = rrow[u6 * 2 + 1];
            float x0 = base[d0], x1 = base[d0 + 1];
            y0 = x0 * c - x1 * s;  y1 = x1 * c + x0 * s;
        } else {
            d0 = 64 + (u6 - 32) * 2;
            y0 = base[d0];  y1 = base[d0 + 1];
        }
        int kvh = h >> 4, g = h & 15;
        size_t o = ((size_t)((b * 2 + kvh) * 16 + g) * 1024 + qq) * 128 + d0;
        *(unsigned*)(qbf + o) = pk2(y0 * scale, y1 * scale);
    }
    {
        int u = tid, h2 = u >> 6, u6 = u & 63;
        const float* base = mrow + 4096 + h2 * 128;
        float y0, y1; int d0;
        if (u6 < 32) {
            d0 = u6 * 2;
            float c = rrow[u6 * 2], s = rrow[u6 * 2 + 1];
            float x0 = base[d0], x1 = base[d0 + 1];
            y0 = x0 * c - x1 * s;  y1 = x1 * c + x0 * s;
        } else {
            d0 = 64 + (u6 - 32) * 2;
            y0 = base[d0];  y1 = base[d0 + 1];
        }
        size_t o = ((size_t)(b * 2 + h2) * 2048 + (1024 + qq)) * 128 + d0;
        *(unsigned*)(kbf + o) = pk2(y0, y1);
    }
    {
        int u = tid, h2 = u >> 6;
        int d0 = (u & 63) * 2;
        const float* base = mrow + 4096 + 256 + h2 * 128;
        size_t rb = (size_t)(b * 2 + h2) * 128;
        vtb[(rb + d0)     * 2048 + 1024 + qq] = f2bf(base[d0]);
        vtb[(rb + d0 + 1) * 2048 + 1024 + qq] = f2bf(base[d0 + 1]);
    }
}

// ---------------------------------------------------------------------------
__global__ __launch_bounds__(128) void gather_hist_k(const float* __restrict__ ck,
                                                     const int* __restrict__ boff,
                                                     bf16_t* __restrict__ kbf)
{
    int r   = blockIdx.x;           // b*2048 + kvh*1024 + s
    int d   = threadIdx.x;
    int s   = r & 1023;
    int kvh = (r >> 10) & 1;
    int b   = r >> 11;
    int blk = boff[b * 32 + (s >> 6)];
    size_t src = (((size_t)blk * 64 + (s & 63)) * 2 + kvh) * 128 + d;
    kbf[((size_t)(b * 2 + kvh) * 2048 + s) * 128 + d] = f2bf(ck[src]);
}

__global__ __launch_bounds__(128) void gather_hist_vt(const float* __restrict__ cv,
                                                      const int* __restrict__ boff,
                                                      bf16_t* __restrict__ vtb)
{
    int r   = blockIdx.x;           // b*256 + kvh*128 + d
    int d   = r & 127;
    int kvh = (r >> 7) & 1;
    int b   = r >> 8;
    bf16_t* dst = vtb + ((size_t)(b * 2 + kvh) * 128 + d) * 2048;
    for (int s = threadIdx.x; s < 1024; s += 128) {
        int blk = boff[b * 32 + (s >> 6)];
        size_t src = (((size_t)blk * 64 + (s & 63)) * 2 + kvh) * 128 + d;
        dst[s] = f2bf(cv[src]);
    }
}

// ---------------------------------------------------------------------------
// Flash attention: one block per (b, kvh, g, 128-row q tile). 8 waves x 16 rows.
// K/V^T chunks double-buffered via GLOBAL_LOAD_ASYNC_TO_LDS_B128 (ASYNCcnt).
// Output ctx stored as bf16 (feeds gemm_abf16 directly).
// ---------------------------------------------------------------------------
__global__ __launch_bounds__(256) void attn(const bf16_t* __restrict__ qbf,
                                            const bf16_t* __restrict__ kbf,
                                            const bf16_t* __restrict__ vtb,
                                            bf16_t* __restrict__ ctx)
{
    __shared__ bf16_t Qs[128 * 128];     // 32 KB
    __shared__ bf16_t Ks[2 * 64 * 128];  // 32 KB (double buffer)
    __shared__ bf16_t Vt[2 * 128 * 64];  // 32 KB (double buffer, [d][s])
    __shared__ bf16_t Ps[8][16 * 64];    // 16 KB (per-wave P tiles)

    const int bid  = blockIdx.x;
    const int qt   = bid & 7;            // 8 q-tiles of 128 rows
    const int g    = (bid >> 3) & 15;
    const int kvh  = (bid >> 7) & 1;
    const int b    = bid >> 8;
    const int tid  = threadIdx.x;
    const int wave = tid >> 5;
    const int lane = tid & 31;
    const int lh   = lane >> 4;
    const int ln   = lane & 15;

    // Async-load Q tile 128x128 bf16
    {
        int row = tid >> 1;
        int cb  = (tid & 1) * 64;
        const bf16_t* gq = qbf +
            (((size_t)((b * 2 + kvh) * 16 + g) * 1024) + qt * 128 + row) * 128 + cb;
        unsigned lq = (unsigned)(size_t)(Qs + row * 128 + cb);
        for (int j = 0; j < 8; ++j) async_ld_b128(lq + j * 16, gq + j * 8);
    }

    const size_t kbase  = (size_t)(b * 2 + kvh) * 2048 * 128;   // kbf [s][d]
    const size_t vtbase = (size_t)(b * 2 + kvh) * 128 * 2048;   // vtb [d][s]
    const int nch = 18 + 2 * qt;

    const int rk  = tid >> 2, ck_ = (tid & 3) * 32;   // K: 64 rows x 128
    const int rv  = tid >> 1, cv_ = (tid & 1) * 32;   // Vt: 128 rows x 64
    const bf16_t* gk0 = kbf + kbase + (size_t)rk * 128 + ck_;
    const bf16_t* gv0 = vtb + vtbase + (size_t)rv * 2048 + cv_;
    const unsigned lk0 = (unsigned)(size_t)(Ks + rk * 128 + ck_);
    const unsigned lv0 = (unsigned)(size_t)(Vt + rv * 64 + cv_);

    auto issue_chunk = [&](int c) {
        const bf16_t* gk = gk0 + (size_t)c * 8192;  // 64*128
        const bf16_t* gv = gv0 + c * 64;
        unsigned lk = lk0 + (c & 1) * 16384;
        unsigned lv = lv0 + (c & 1) * 16384;
        for (int j = 0; j < 4; ++j) async_ld_b128(lk + j * 16, gk + j * 8);
        for (int j = 0; j < 4; ++j) async_ld_b128(lv + j * 16, gv + j * 8);
        __builtin_prefetch(gk + 8192, 0, 3);   // chunk c+1 into near cache
        __builtin_prefetch(gv + 64, 0, 3);
    };
    issue_chunk(0);      // Q copies + chunk0 copies outstanding
    wait_async0();
    __syncthreads();

    // Preload Q A-fragments (constant across chunks)
    V16u aQ[4];
    {
        const bf16_t* pa = Qs + (wave * 16 + ln) * 128;
        int kb = lh * 8;
        for (int ks = 0; ks < 4; ++ks) {
            aQ[ks].h[0] = *(const v8bf*)(pa + ks * 32 + kb);
            aQ[ks].h[1] = *(const v8bf*)(pa + ks * 32 + 16 + kb);
        }
    }

    float mst[8], lst[8];
    v8f o[8];
    for (int r = 0; r < 8; ++r) { mst[r] = -3.0e38f; lst[r] = 0.0f; }
    for (int dt = 0; dt < 8; ++dt)
        for (int r = 0; r < 8; ++r) o[dt][r] = 0.0f;

    const int qlo = 1024 + qt * 128 + wave * 16;   // wave's lowest q position

    for (int c = 0; c < nch; ++c) {
        if (c) { wait_async0(); __syncthreads(); }
        if (c + 1 < nch) issue_chunk(c + 1);       // overlap next chunk DMA

        if (c * 64 > qlo + 15) continue;           // whole wave masked (uniform)

        const bf16_t* Kc = Ks + (c & 1) * 8192;
        const bf16_t* Vc = Vt + (c & 1) * 8192;

        // S = Q * K^T
        v8f sa[4];
        for (int nt = 0; nt < 4; ++nt)
            for (int r = 0; r < 8; ++r) sa[nt][r] = 0.0f;
        for (int nt = 0; nt < 4; ++nt) {
            const bf16_t* pb = Kc + (nt * 16 + ln) * 128;
            for (int ks = 0; ks < 4; ++ks) {
                V16u bK;
                bK.h[0] = *(const v8bf*)(pb + ks * 32 + lh * 8);
                bK.h[1] = *(const v8bf*)(pb + ks * 32 + 16 + lh * 8);
                sa[nt] = __builtin_amdgcn_wmma_f32_16x16x32_bf16(
                    false, aQ[ks].v, false, bK.v, (short)0, sa[nt], false, false);
            }
        }

        // Causal mask (wave-uniform branch)
        if (c * 64 + 63 > qlo) {
            for (int nt = 0; nt < 4; ++nt)
                for (int r = 0; r < 8; ++r) {
                    int s  = c * 64 + nt * 16 + ln;
                    int qp = qlo + r + 8 * lh;
                    if (s > qp) sa[nt][r] = -1.0e30f;
                }
        }

        // Online softmax
        float alpha[8];
        for (int r = 0; r < 8; ++r) {
            float cm = sa[0][r];
            for (int nt = 1; nt < 4; ++nt) cm = fmaxf(cm, sa[nt][r]);
            for (int off = 1; off < 16; off <<= 1)
                cm = fmaxf(cm, __shfl_xor(cm, off, 32));
            float nm = fmaxf(mst[r], cm);
            alpha[r] = __expf(mst[r] - nm);
            mst[r] = nm;
        }
        for (int r = 0; r < 8; ++r) {
            float rs = 0.0f;
            for (int nt = 0; nt < 4; ++nt) {
                float p = __expf(sa[nt][r] - mst[r]);
                sa[nt][r] = p;
                rs += p;
            }
            for (int off = 1; off < 16; off <<= 1)
                rs += __shfl_xor(rs, off, 32);
            lst[r] = lst[r] * alpha[r] + rs;
        }
        for (int dt = 0; dt < 8; ++dt)
            for (int r = 0; r < 8; ++r) o[dt][r] *= alpha[r];

        // P -> LDS (per-wave region) for WMMA A re-fragmentation
        for (int nt = 0; nt < 4; ++nt)
            for (int r = 0; r < 8; ++r)
                Ps[wave][(r + 8 * lh) * 64 + nt * 16 + ln] = f2bf(sa[nt][r]);
        asm volatile("s_wait_dscnt 0" ::: "memory");

        // O += P * V
        V16u aP[2];
        {
            const bf16_t* pp = &Ps[wave][ln * 64];
            int kb = lh * 8;
            for (int k2 = 0; k2 < 2; ++k2) {
                aP[k2].h[0] = *(const v8bf*)(pp + k2 * 32 + kb);
                aP[k2].h[1] = *(const v8bf*)(pp + k2 * 32 + 16 + kb);
            }
        }
        for (int dt = 0; dt < 8; ++dt) {
            const bf16_t* pv = Vc + (dt * 16 + ln) * 64;
            for (int k2 = 0; k2 < 2; ++k2) {
                V16u bV;
                bV.h[0] = *(const v8bf*)(pv + k2 * 32 + lh * 8);
                bV.h[1] = *(const v8bf*)(pv + k2 * 32 + 16 + lh * 8);
                o[dt] = __builtin_amdgcn_wmma_f32_16x16x32_bf16(
                    false, aP[k2].v, false, bV.v, (short)0, o[dt], false, false);
            }
        }
    }

    // Normalize + store ctx (bf16) [sq][h*128 + d]
    int h = kvh * 16 + g;
    for (int r = 0; r < 8; ++r) {
        float inv = 1.0f / lst[r];
        int m = r + 8 * lh;
        size_t sq = (size_t)b * 1024 + qt * 128 + wave * 16 + m;
        for (int dt = 0; dt < 8; ++dt)
            ctx[sq * 4096 + h * 128 + dt * 16 + ln] = f2bf(o[dt][r] * inv);
    }
}

// ---------------------------------------------------------------------------
extern "C" void kernel_launch(void* const* d_in, const int* in_sizes, int n_in,
                              void* d_out, int out_size, void* d_ws, size_t ws_size,
                              hipStream_t stream)
{
    const float* hidden = (const float*)d_in[0];
    const float* rope   = (const float*)d_in[1];
    const float* ck     = (const float*)d_in[2];
    const float* cv     = (const float*)d_in[3];
    const int*   boff   = (const int*)d_in[4];
    const float* wqkv   = (const float*)d_in[9];
    const float* bqkv   = (const float*)d_in[10];
    const float* wdense = (const float*)d_in[11];
    float* out = (float*)d_out;

    char* ws = (char*)d_ws;
    const size_t OFF_MIXED = 0;                               // 4096*4608*4  = 75497472
    const size_t OFF_CTX   = OFF_MIXED + 75497472;            // 4096*4096*2  = 33554432 (bf16)
    const size_t OFF_QBF   = OFF_CTX   + 33554432;            // 4096*4096*2  = 33554432
    const size_t OFF_KBF   = OFF_QBF   + 33554432;            // 4*2*2048*128*2 = 4194304
    const size_t OFF_VTB   = OFF_KBF   + 4194304;

    float*  mixed = (float*)(ws + OFF_MIXED);
    bf16_t* ctxbf = (bf16_t*)(ws + OFF_CTX);
    bf16_t* qbf   = (bf16_t*)(ws + OFF_QBF);
    bf16_t* kbf   = (bf16_t*)(ws + OFF_KBF);
    bf16_t* vtb   = (bf16_t*)(ws + OFF_VTB);

    // 1) mixed = hidden @ w_qkv^T + b_qkv
    gemm_bf16<<<dim3(4608 / 128, 4096 / 128), 256, 0, stream>>>(
        hidden, wqkv, bqkv, mixed, 4096, 4608, 4096);
    // 2) RoPE + reorder Q, new K; V transposed
    prep_qkv<<<4096, 128, 0, stream>>>(mixed, rope, qbf, kbf, vtb);
    // 3) Gather history KV from paged cache
    gather_hist_k<<<8192, 128, 0, stream>>>(ck, boff, kbf);
    gather_hist_vt<<<1024, 128, 0, stream>>>(cv, boff, vtb);
    // 4) Attention -> ctx (bf16)
    attn<<<1024, 256, 0, stream>>>(qbf, kbf, vtb, ctxbf);
    // 5) out = ctx @ w_dense^T  (A already bf16, async-staged)
    gemm_abf16<<<dim3(4096 / 128, 4096 / 128), 256, 0, stream>>>(
        ctxbf, wdense, out, 4096, 4096, 4096);
}
